// SimpleBlock_35725537968379
// MI455X (gfx1250) — matrix operand order repaired and checked by reading, not compiled, not run on hardware
//
#include <hip/hip_runtime.h>

// ---------------------------------------------------------------------------
// EKPConv_v2 forward for MI455X (gfx1250, wave32, WMMA)
// ---------------------------------------------------------------------------

typedef _Float16 v16h __attribute__((ext_vector_type(16)));
typedef _Float16 v8h  __attribute__((ext_vector_type(8)));
typedef float    v8f  __attribute__((ext_vector_type(8)));

#define N_Q    20000
#define N_S    20000
#define NEIGH  30
#define KPTS   15
#define NQn    4
#define NSn    4
#define LRF_C  32
#define CIN    128
#define COUT   256
#define KDIM   3840        // 15 * 256 reduction dimension of the final GEMM
#define QB     16          // queries per block (== one WMMA M-tile)
#define SWS    3848        // sW row stride in halfs (3840 + 8 pad -> bank-conflict-free)

// ---- dynamic LDS layout (all offsets 16B aligned) -------------------------
constexpr int OFF_SW   = 0;                       // _Float16[QB][SWS]   123136 B
constexpr int OFF_WL   = OFF_SW   + QB * SWS * 2; // float[36*32]          4608 B
constexpr int OFF_KP   = OFF_WL   + 1152 * 4;     // float[48]              192 B
constexpr int OFF_BL   = OFF_KP   + 48 * 4;       // float[32]              128 B
constexpr int OFF_QL   = OFF_BL   + 32 * 4;       // float[36]              144 B
constexpr int OFF_QP   = OFF_QL   + 36 * 4;       // float[4]                16 B
constexpr int OFF_NI   = OFF_QP   + 4 * 4;        // int[32]                128 B
constexpr int OFF_AW   = OFF_NI   + 32 * 4;       // float[4*15*30]        7200 B
constexpr int OFF_ALRF = OFF_AW   + 1800 * 4;     // float[30*4*36]       17280 B
constexpr int OFF_NX   = OFF_ALRF + 4320 * 4;     // float[4*30*64]       30720 B
constexpr int LDS_TOTAL = OFF_NX  + 7680 * 4;     // = 183552 B (< 320 KB / WG)

// ---------------------------------------------------------------------------
// Kernel 1: repack weights[k][i][o] (f32) -> Bt[o][k*256+i] (f16, column-major
// for the GEMM so each lane's B fragment is 16 contiguous halfs).
// ---------------------------------------------------------------------------
__global__ __launch_bounds__(256)
void prep_bt(const float* __restrict__ w, _Float16* __restrict__ Bt)
{
    const int t = blockIdx.x * 256 + threadIdx.x;
    if (t >= KDIM * COUT) return;
    const int o = t & 255;       // output channel
    const int j = t >> 8;        // reduction index = k*256 + i
    Bt[(size_t)o * KDIM + j] = (_Float16)w[(size_t)j * COUT + o];
}

// ---------------------------------------------------------------------------
// Kernel 2: fully fused EKPConv for 16 queries per block.
// Phase 1 (f32 VALU): gathers + KP weights + LRF features + aw@nx -> LDS f16.
// Phase 2 (WMMA):     out[16x256] = sW[16x3840] x Bt^T, bias + LeakyReLU.
// ---------------------------------------------------------------------------
__global__ __launch_bounds__(256)
void ekp_fused(const float* __restrict__ q_pts, const float* __restrict__ s_pts,
               const int*   __restrict__ nb_g,  const float* __restrict__ x,
               const float* __restrict__ q_lrf, const float* __restrict__ s_lrf,
               const float* __restrict__ kp_g,  const float* __restrict__ W_lrf,
               const float* __restrict__ b_lrf, const float* __restrict__ bias,
               const _Float16* __restrict__ Bt, float* __restrict__ out)
{
    extern __shared__ char smem[];
    _Float16* sW   = (_Float16*)(smem + OFF_SW);
    float*    sWl  = (float*)(smem + OFF_WL);
    float*    sKP  = (float*)(smem + OFF_KP);
    float*    sBl  = (float*)(smem + OFF_BL);
    float*    sQL  = (float*)(smem + OFF_QL);
    float*    sQP  = (float*)(smem + OFF_QP);
    int*      sNI  = (int*)  (smem + OFF_NI);
    float*    sAW  = (float*)(smem + OFF_AW);
    float*    sALR = (float*)(smem + OFF_ALRF);
    float*    sNX  = (float*)(smem + OFF_NX);

    const int tid = threadIdx.x;

    // one-time block-constant loads
    for (int u = tid; u < 36 * 32; u += 256) sWl[u] = W_lrf[u];
    for (int u = tid; u < KPTS * 3; u += 256) sKP[u] = kp_g[u];
    if (tid < 32) sBl[tid] = b_lrf[tid];
    __syncthreads();

    // ---------------- phase 1: build sW[qi][3840] per query ----------------
    for (int qi = 0; qi < QB; ++qi) {
        const int n = blockIdx.x * QB + qi;
        if (tid < 36)    sQL[tid] = q_lrf[(size_t)n * 36 + tid];
        if (tid < 3)     sQP[tid] = q_pts[(size_t)n * 3 + tid];
        if (tid < NEIGH) sNI[tid] = nb_g[(size_t)n * NEIGH + tid];
        __syncthreads();

        // (b) kernel-point influence: aw[q][k][m] = clip(1 - |aligned-kp|/ext)
        if (tid < NQn * NEIGH) {
            const int q = tid / NEIGH, m = tid % NEIGH;
            const int idx = sNI[m];
            float p0, p1, p2;
            if (idx >= N_S) { p0 = p1 = p2 = 1e6f; }
            else { p0 = s_pts[idx*3]; p1 = s_pts[idx*3+1]; p2 = s_pts[idx*3+2]; }
            const float d0 = p0 - sQP[0], d1 = p1 - sQP[1], d2 = p2 - sQP[2];
            const float* ql = &sQL[q * 9];
            const float a0 = d0*ql[0] + d1*ql[3] + d2*ql[6];
            const float a1 = d0*ql[1] + d1*ql[4] + d2*ql[7];
            const float a2 = d0*ql[2] + d1*ql[5] + d2*ql[8];
            #pragma unroll
            for (int k = 0; k < KPTS; ++k) {
                const float ex = a0 - sKP[k*3], ey = a1 - sKP[k*3+1], ez = a2 - sKP[k*3+2];
                const float w  = 1.0f - sqrtf(ex*ex + ey*ey + ez*ez) / 0.05f;
                sAW[(q*KPTS + k)*NEIGH + m] = fmaxf(w, 0.0f);
            }
        }
        // (c) aligned LRFs: alr[m][q][s*9+i*3+j] = sum_kk ql[q][kk][i]*slrf[idx][s][kk][j]
        for (int e = tid; e < NEIGH * NQn * 36; e += 256) {
            const int m = e / (NQn * 36); int r = e % (NQn * 36);
            const int q = r / 36, u = r % 36;
            const int s = u / 9, v = u % 9, i = v / 3, j = v % 3;
            const int idx = sNI[m];
            float val = 0.0f;
            if (idx < N_S) {
                const float* sl = &s_lrf[(size_t)(idx*NSn + s) * 9 + j];
                const float* ql = &sQL[q*9 + i];
                val = ql[0]*sl[0] + ql[3]*sl[3] + ql[6]*sl[6];
            }
            sALR[(m*NQn + q)*36 + u] = val;
        }
        __syncthreads();

        // (d) neighbor features: nx[q][m][c] = [ gathered x | lrf linear ]
        for (int e = tid; e < NQn * NEIGH * 64; e += 256) {
            const int q = e / (NEIGH * 64); int r = e % (NEIGH * 64);
            const int m = r / 64, c = r % 64;
            const int idx = sNI[m];
            float val = 0.0f;
            if (idx < N_S) {
                if (c < LRF_C) {
                    val = x[(size_t)idx * CIN + q * LRF_C + c];
                } else {
                    const int cc = c - LRF_C;
                    float acc = sBl[cc];
                    const float* al = &sALR[(m*NQn + q)*36];
                    #pragma unroll
                    for (int u = 0; u < 36; ++u) acc += al[u] * sWl[u*LRF_C + cc];
                    val = acc;
                }
            }
            sNX[(q*NEIGH + m)*64 + c] = val;
        }
        __syncthreads();

        // (e) weighted[q][k][c] = sum_m aw[q][k][m]*nx[q][m][c]; pack f16 GEMM row
        for (int e = tid; e < NQn * KPTS * 64; e += 256) {
            const int q = e / (KPTS * 64); int r = e % (KPTS * 64);
            const int k = r / 64, c = r % 64;
            const float* aw = &sAW[(q*KPTS + k)*NEIGH];
            const float* nx = &sNX[(q*NEIGH)*64 + c];
            float acc = 0.0f;
            #pragma unroll
            for (int m = 0; m < NEIGH; ++m) acc += aw[m] * nx[m*64];
            sW[qi*SWS + k*256 + q*64 + c] = (_Float16)acc;   // j = k*256 + q*64 + c
        }
        __syncthreads();
    }

    // ---------------- phase 2: WMMA GEMM (uniform flow, EXEC all-1s) -------
    const int lane = tid & 31;
    const int wv   = tid >> 5;          // 8 waves x 32 cols = 256 output cols
    const int hh   = lane >> 4;         // K-half selector (this code's A/B order k = 16*hh + s)
    const int rr   = lane & 15;         // A row / B column within tile
    const _Float16* a_row = sW + rr * SWS;
    const _Float16* b0p = Bt + (size_t)(wv*32 + rr)      * KDIM;
    const _Float16* b1p = Bt + (size_t)(wv*32 + 16 + rr) * KDIM;

    v8f c0 = {}, c1 = {};
    for (int kb = 0; kb < KDIM; kb += 32) {
        // A fragment (16-bit A 16x32): per-lane 16 contiguous K halfs at hh*16
        v8h alo = *(const v8h*)(a_row + kb + hh*16);
        v8h ahi = *(const v8h*)(a_row + kb + hh*16 + 8);
        v16h A = __builtin_shufflevector(alo, ahi,
                 0,1,2,3,4,5,6,7,8,9,10,11,12,13,14,15);
        // B fragments (16-bit B 32x16): lane=column, 16 contiguous K halfs
        const int krun = kb + hh*16;
        v8h b0l = *(const v8h*)(b0p + krun);
        v8h b0h = *(const v8h*)(b0p + krun + 8);
        v16h B0 = __builtin_shufflevector(b0l, b0h,
                 0,1,2,3,4,5,6,7,8,9,10,11,12,13,14,15);
        v8h b1l = *(const v8h*)(b1p + krun);
        v8h b1h = *(const v8h*)(b1p + krun + 8);
        v16h B1 = __builtin_shufflevector(b1l, b1h,
                 0,1,2,3,4,5,6,7,8,9,10,11,12,13,14,15);
        c0 = __builtin_amdgcn_wmma_f32_16x16x32_f16(false, A, false, B0,
                                                    (short)0, c0, false, false);
        c1 = __builtin_amdgcn_wmma_f32_16x16x32_f16(false, A, false, B1,
                                                    (short)0, c1, false, false);
    }

    // C/D layout: VGPR v, lane l -> M = v + 8*(l>>4), Ncol = l&15
    const int n0 = blockIdx.x * QB;
    #pragma unroll
    for (int v = 0; v < 8; ++v) {
        const int row  = n0 + v + hh*8;
        const int col0 = wv*32 + rr;
        const int col1 = col0 + 16;
        float o0 = c0[v] + bias[col0];
        float o1 = c1[v] + bias[col1];
        out[(size_t)row * COUT + col0] = (o0 >= 0.0f) ? o0 : 0.1f * o0;
        out[(size_t)row * COUT + col1] = (o1 >= 0.0f) ? o1 : 0.1f * o1;
    }
}

// ---------------------------------------------------------------------------
extern "C" void kernel_launch(void* const* d_in, const int* in_sizes, int n_in,
                              void* d_out, int out_size, void* d_ws, size_t ws_size,
                              hipStream_t stream)
{
    const float* q_pts = (const float*)d_in[0];
    const float* s_pts = (const float*)d_in[1];
    const int*   nb    = (const int*)  d_in[2];
    const float* x     = (const float*)d_in[3];
    const float* q_lrf = (const float*)d_in[4];
    const float* s_lrf = (const float*)d_in[5];
    const float* kp    = (const float*)d_in[6];
    const float* wts   = (const float*)d_in[7];
    const float* W_lrf = (const float*)d_in[8];
    const float* b_lrf = (const float*)d_in[9];
    const float* bias  = (const float*)d_in[10];
    float*       out   = (float*)d_out;
    _Float16*    Bt    = (_Float16*)d_ws;     // 256*3840*2 B = ~1.97 MB

    prep_bt<<<(KDIM * COUT + 255) / 256, 256, 0, stream>>>(wts, Bt);

    static_assert(LDS_TOTAL <= 320 * 1024, "LDS budget");
    (void)hipFuncSetAttribute(reinterpret_cast<const void*>(ekp_fused),
                              hipFuncAttributeMaxDynamicSharedMemorySize,
                              LDS_TOTAL);
    ekp_fused<<<N_Q / QB, 256, LDS_TOTAL, stream>>>(
        q_pts, s_pts, nb, x, q_lrf, s_lrf, kp, W_lrf, b_lrf, bias, Bt, out);
}
